// Net_18322330484803
// MI455X (gfx1250) — compile-verified, matrix operand order checked by reading, not run on hardware
//
#include <hip/hip_runtime.h>
#include <hip/hip_bf16.h>
#include <stdint.h>
#include <stddef.h>

#define NCAM 4
#define NPTS 16384

typedef __attribute__((ext_vector_type(16))) __bf16 bf16x16;
typedef __attribute__((ext_vector_type(8)))  float  f32x8;

__device__ __forceinline__ float bf2f(__bf16 h) {
  unsigned short u = __builtin_bit_cast(unsigned short, h);
  unsigned int b = ((unsigned int)u) << 16;
  return __builtin_bit_cast(float, b);
}
__device__ __forceinline__ __bf16 f2bf(float f) {
  unsigned int b = __builtin_bit_cast(unsigned int, f);
  b += 0x7FFFu + ((b >> 16) & 1u);       // round-to-nearest-even
  unsigned short u = (unsigned short)(b >> 16);
  return __builtin_bit_cast(__bf16, u);
}

// ---------------------------------------------------------------------------
// conv3x3 (SAME) + per-channel affine (+ optional ReLU), NCHW fp32
// ---------------------------------------------------------------------------
__global__ void conv3x3_k(const float* __restrict__ src, const float* __restrict__ wgt,
                          const float* __restrict__ bias, const float* __restrict__ scl,
                          const float* __restrict__ off, float* __restrict__ dst,
                          int Ci, int Co, int Hs, int Ws, int relu) {
  int idx = blockIdx.x * blockDim.x + threadIdx.x;
  int total = NCAM * Co * Hs * Ws;
  if (idx >= total) return;
  int w  = idx % Ws;
  int h  = (idx / Ws) % Hs;
  int co = (idx / (Ws * Hs)) % Co;
  int cam = idx / (Ws * Hs * Co);

  float acc = bias[co];
  for (int ci = 0; ci < Ci; ++ci) {
    const float* s = src + ((size_t)(cam * Ci + ci) * Hs) * Ws;
    const float* wp = wgt + ((size_t)(co * Ci + ci) * 9);
    #pragma unroll
    for (int dh = -1; dh <= 1; ++dh) {
      int hh = h + dh;
      if (hh < 0 || hh >= Hs) continue;
      #pragma unroll
      for (int dw = -1; dw <= 1; ++dw) {
        int ww = w + dw;
        if (ww < 0 || ww >= Ws) continue;
        acc += s[(size_t)hh * Ws + ww] * wp[(dh + 1) * 3 + (dw + 1)];
      }
    }
  }
  float y = acc * scl[co] + off[co];
  if (relu) y = y > 0.f ? y : 0.f;
  dst[idx] = y;
}

// ---------------------------------------------------------------------------
// 2x2 maxpool, NCHW fp32.  Ho, Wo are OUTPUT dims.
// ---------------------------------------------------------------------------
__global__ void maxpool2_k(const float* __restrict__ src, float* __restrict__ dst,
                           int C, int Ho, int Wo) {
  int idx = blockIdx.x * blockDim.x + threadIdx.x;
  int total = NCAM * C * Ho * Wo;
  if (idx >= total) return;
  int wo = idx % Wo;
  int ho = (idx / Wo) % Ho;
  int cc = (idx / (Wo * Ho)) % C;
  int cam = idx / (Wo * Ho * C);
  const float* s = src + (((size_t)(cam * C + cc) * (2 * Ho)) + 2 * ho) * (size_t)(2 * Wo) + 2 * wo;
  float m = fmaxf(fmaxf(s[0], s[1]), fmaxf(s[2 * Wo], s[2 * Wo + 1]));
  dst[idx] = m;
}

// ---------------------------------------------------------------------------
// Bilinear sample (exact reference formula) -> bf16 feature matrix
// dst: [cam*NPTS, 1024], writes channels [chOff, chOff+C)
// ---------------------------------------------------------------------------
__global__ void bilinear_sample_k(const float* __restrict__ feat, const float* __restrict__ pts,
                                  __bf16* __restrict__ dst, int C, int Hs, int Ws,
                                  int chOff, float coordScale) {
  int idx = blockIdx.x * blockDim.x + threadIdx.x;
  int total = NCAM * NPTS * C;
  if (idx >= total) return;
  int c   = idx % C;
  int pt  = (idx / C) % NPTS;
  int cam = idx / (C * NPTS);

  float px = pts[((size_t)cam * NPTS + pt) * 2 + 0] * coordScale;
  float py = pts[((size_t)cam * NPTS + pt) * 2 + 1] * coordScale;
  float xc_f = ceilf(px), xf_f = xc_f - 1.0f;
  float yc_f = ceilf(py), yf_f = yc_f - 1.0f;
  float inside = (px <= (float)(Hs - 1) && px >= 0.f &&
                  py <= (float)(Ws - 1) && py >= 0.f) ? 1.f : 0.f;
  int xc = (int)(inside * xc_f), xf = (int)(inside * xf_f);
  int yc = (int)(inside * yc_f), yf = (int)(inside * yf_f);

  const float* f = feat + ((size_t)cam * C + c) * (size_t)Hs * Ws;
  float g_cc = f[(size_t)xc * Ws + yc];
  float g_fc = f[(size_t)xf * Ws + yc];
  float g_cf = f[(size_t)xc * Ws + yf];
  float g_ff = f[(size_t)xf * Ws + yf];
  float wxf = xf_f - px, wxc = px - xc_f;
  float wyf = yf_f - py, wyc = py - yc_f;
  float val = g_cc * wxf * wyf + g_fc * wxc * wyf + g_cf * wxf * wyc + g_ff * wxc * wyc;
  dst[((size_t)cam * NPTS + pt) * 1024 + chOff + c] = f2bf(val);
}

// zero-fill pad channels 1016..1023 of the feature matrix
__global__ void pad_feats_k(__bf16* __restrict__ dst) {
  int idx = blockIdx.x * blockDim.x + threadIdx.x;
  int total = NCAM * NPTS * 8;
  if (idx >= total) return;
  int c = 1016 + (idx % 8);
  int row = idx / 8;
  dst[(size_t)row * 1024 + c] = f2bf(0.f);
}

// convert fp32 weight [N,K] -> bf16 [N,Kp] (zero padded K)
__global__ void wconv_k(const float* __restrict__ w, __bf16* __restrict__ o,
                        int N, int K, int Kp) {
  int idx = blockIdx.x * blockDim.x + threadIdx.x;
  int total = N * Kp;
  if (idx >= total) return;
  int k = idx % Kp;
  int n = idx / Kp;
  o[idx] = (k < K) ? f2bf(w[(size_t)n * K + k]) : f2bf(0.f);
}

// ---------------------------------------------------------------------------
// LDS-staged WMMA bf16 GEMM:  out[M,N] = relu(A[M,K] x W[N,K]^T + bias)
//
// Block = 256 threads = 8 waves in a 4(M) x 2(N) grid.
// Block tile: 256(M) x 128(N); per-wave tile: 64x64 = 4x4 WMMA tiles.
// K is consumed in 32-element steps; A/B K-slices are double-buffered in LDS,
// filled with global_load_async_to_lds_b128 and drained with s_wait_asynccnt.
// Fragment addressing follows ISA 7.12.2 (lane=row, lane[4]=K-half).
// ---------------------------------------------------------------------------
union FragU { bf16x16 v; uint4 q[2]; };

#define GBM 256
#define GBN 128
#define GKS 32
#define GLDR 40   // bf16 per LDS row (32 data + 8 pad) -> 80B row, 16B aligned

__global__ void __launch_bounds__(256) gemm_bf16_wmma_lds(
    const __bf16* __restrict__ A, const __bf16* __restrict__ W,
    const float* __restrict__ bias, __bf16* __restrict__ out,
    int M, int N, int K, int relu) {
  __shared__ __align__(16) __bf16 Asm[2][GBM * GLDR];   // 2 x 20480 B
  __shared__ __align__(16) __bf16 Bsm[2][GBN * GLDR];   // 2 x 10240 B

  const int tid  = threadIdx.x;
  const int lane = tid & 31;
  const int wave = tid >> 5;
  const int waveM = wave >> 1;           // 0..3
  const int waveN = wave & 1;            // 0..1
  const int r16 = lane & 15;             // row within 16x16 tile
  const int kh  = (lane >> 4) << 3;      // K-half select: 0 or 8 elements

  const int m0 = blockIdx.y * GBM;
  const int n0 = blockIdx.x * GBN;

  // ---- async copy plan ----
  // A tile: 256 rows x 64B; thread t copies row t (4 x b128 chunks).
  // B tile: 128 rows x 64B; thread t copies 32B of row (t&127) at 32B*(t>>7).
  const int rowA = tid;
  const int rowB = tid & 127;
  const int colB = (tid >> 7) * 16;      // element offset (16 bf16 = 32B)

  const __bf16* gA = A + (size_t)(m0 + rowA) * K;
  const __bf16* gB = W + (size_t)(n0 + rowB) * K + colB;
  const unsigned laA = (unsigned)(uintptr_t)&Asm[0][rowA * GLDR];
  const unsigned laB = (unsigned)(uintptr_t)&Bsm[0][rowB * GLDR + colB];
  const unsigned strideA = (unsigned)(GBM * GLDR * 2);
  const unsigned strideB = (unsigned)(GBN * GLDR * 2);

  auto issue = [&](int buf, int k0) {
    const __bf16* ga = gA + k0;
    const __bf16* gb = gB + k0;
    unsigned la = laA + (unsigned)buf * strideA;
    unsigned lb = laB + (unsigned)buf * strideB;
    // offset: immediate applies to BOTH the global and LDS addresses
    asm volatile("global_load_async_to_lds_b128 %0, %1, off"           :: "v"(la), "v"(ga) : "memory");
    asm volatile("global_load_async_to_lds_b128 %0, %1, off offset:16" :: "v"(la), "v"(ga) : "memory");
    asm volatile("global_load_async_to_lds_b128 %0, %1, off offset:32" :: "v"(la), "v"(ga) : "memory");
    asm volatile("global_load_async_to_lds_b128 %0, %1, off offset:48" :: "v"(la), "v"(ga) : "memory");
    asm volatile("global_load_async_to_lds_b128 %0, %1, off"           :: "v"(lb), "v"(gb) : "memory");
    asm volatile("global_load_async_to_lds_b128 %0, %1, off offset:16" :: "v"(lb), "v"(gb) : "memory");
  };

  f32x8 acc[4][4] = {};
  const int nsteps = K / GKS;
  issue(0, 0);

  #pragma unroll 1
  for (int s = 0; s < nsteps; ++s) {
    asm volatile("s_wait_asynccnt 0x0" ::: "memory");  // our slice has landed
    __syncthreads();                                    // whole tile visible; prev reads done
    if (s + 1 < nsteps) issue((s + 1) & 1, (s + 1) * GKS);

    const __bf16* Ab = Asm[s & 1];
    const __bf16* Bb = Bsm[s & 1];

    bf16x16 bF[4];
    #pragma unroll
    for (int j = 0; j < 4; ++j) {
      const __bf16* p = Bb + (waveN * 64 + j * 16 + r16) * GLDR + kh;
      FragU f;
      f.q[0] = *(const uint4*)p;          // K = kh + 0..7
      f.q[1] = *(const uint4*)(p + 16);   // K = 16 + kh + 0..7
      bF[j] = f.v;
    }
    #pragma unroll
    for (int i = 0; i < 4; ++i) {
      const __bf16* p = Ab + (waveM * 64 + i * 16 + r16) * GLDR + kh;
      FragU f;
      f.q[0] = *(const uint4*)p;
      f.q[1] = *(const uint4*)(p + 16);
      #pragma unroll
      for (int j = 0; j < 4; ++j) {
        acc[i][j] = __builtin_amdgcn_wmma_f32_16x16x32_bf16(
            false, f.v, false, bF[j], (short)0, acc[i][j], false, false);
      }
    }
  }

  // ---- epilogue: C/D layout: VGPR r -> M=r (+8 for lanes 16..31), N = lane&15
  const int mBase = m0 + waveM * 64 + ((lane >> 4) << 3);
  const int nBase = n0 + waveN * 64;
  #pragma unroll
  for (int j = 0; j < 4; ++j) {
    const int n = nBase + j * 16 + r16;
    const float bv = bias[n];
    #pragma unroll
    for (int i = 0; i < 4; ++i) {
      #pragma unroll
      for (int r = 0; r < 8; ++r) {
        float v = acc[i][j][r] + bv;
        if (relu) v = v > 0.f ? v : 0.f;
        out[(size_t)(mBase + i * 16 + r) * N + n] = f2bf(v);
      }
    }
  }
}

// ---------------------------------------------------------------------------
// camera pooling: y3 [4, NPTS, 1024] bf16 -> pooled [NPTS, 2048] bf16
// ---------------------------------------------------------------------------
__global__ void pool_cams_k(const __bf16* __restrict__ y, __bf16* __restrict__ out) {
  int idx = blockIdx.x * blockDim.x + threadIdx.x;
  int total = NPTS * 1024;
  if (idx >= total) return;
  int c = idx % 1024;
  int n = idx / 1024;
  float mx = -3.4e38f, sm = 0.f;
  #pragma unroll
  for (int cam = 0; cam < NCAM; ++cam) {
    float v = bf2f(y[((size_t)cam * NPTS + n) * 1024 + c]);
    mx = fmaxf(mx, v);
    sm += v;
  }
  out[(size_t)n * 2048 + c]        = f2bf(mx);
  out[(size_t)n * 2048 + 1024 + c] = f2bf(sm * 0.25f);
}

// final fc6: [NPTS,128] bf16 x [2,128] fp32 + bias -> d_out fp32 [NPTS,2]
__global__ void fc6_k(const __bf16* __restrict__ y, const float* __restrict__ w,
                      const float* __restrict__ b, float* __restrict__ out) {
  int n = blockIdx.x * blockDim.x + threadIdx.x;
  if (n >= NPTS) return;
  float a0 = b[0], a1 = b[1];
  #pragma unroll 4
  for (int k = 0; k < 128; ++k) {
    float v = bf2f(y[(size_t)n * 128 + k]);
    a0 += v * w[k];
    a1 += v * w[128 + k];
  }
  out[n * 2 + 0] = a0;
  out[n * 2 + 1] = a1;
}

// ---------------------------------------------------------------------------
static inline int cdiv(int a, int b) { return (a + b - 1) / b; }

extern "C" void kernel_launch(void* const* d_in, const int* in_sizes, int n_in,
                              void* d_out, int out_size, void* d_ws, size_t ws_size,
                              hipStream_t stream) {
  (void)in_sizes; (void)n_in; (void)out_size; (void)ws_size;
  // ---- inputs (jax pytree order: x, projection_points, params{sorted keys}) ----
  const float* x   = (const float*)d_in[0];
  const float* pts = (const float*)d_in[1];
  const float* b1o = (const float*)d_in[2];
  const float* b1s = (const float*)d_in[3];
  const float* c1b = (const float*)d_in[4];
  const float* c1w = (const float*)d_in[5];
  const float* fc1b = (const float*)d_in[6];
  const float* fc1w = (const float*)d_in[7];
  const float* fc2b = (const float*)d_in[8];
  const float* fc2w = (const float*)d_in[9];
  const float* fc3b = (const float*)d_in[10];
  const float* fc3w = (const float*)d_in[11];
  const float* fc4b = (const float*)d_in[12];
  const float* fc4w = (const float*)d_in[13];
  const float* fc5b = (const float*)d_in[14];
  const float* fc5w = (const float*)d_in[15];
  const float* fc6b = (const float*)d_in[16];
  const float* fc6w = (const float*)d_in[17];
  // scales[s] = (w1,b1,s1,o1,w2,b2,s2,o2) at d_in[18 + 8*s + j]
  #define SC(s, j) ((const float*)d_in[18 + 8 * (s) + (j)])

  // ---- workspace layout ----
  char* base = (char*)d_ws;
  const size_t SZ_R = (size_t)NCAM * NPTS * 1024 * 2;  // 134 MB
  __bf16* feats = (__bf16*)base;                       // R0
  char*   R1    = base + SZ_R;
  char*   WBp   = base + 2 * SZ_R;
  float* F0 = (float*)(R1);                            // conv feature maps
  float* F1 = (float*)(R1 + ((size_t)36 << 20));
  float* PB = (float*)(R1 + ((size_t)56 << 20));
  float* MB = (float*)(R1 + ((size_t)66 << 20));
  __bf16* wb1 = (__bf16*)WBp;
  __bf16* wb2 = wb1 + (size_t)512 * 1024;
  __bf16* wb3 = wb2 + (size_t)512 * 512;
  __bf16* wb4 = wb3 + (size_t)1024 * 512;
  __bf16* wb5 = wb4 + (size_t)512 * 2048;

  const int B = 256;

  // ---- 1) weight conversion to bf16 (K padded for fc1) ----
  wconv_k<<<cdiv(512 * 1024, B), B, 0, stream>>>(fc1w, wb1, 512, 1016, 1024);
  wconv_k<<<cdiv(512 * 512,  B), B, 0, stream>>>(fc2w, wb2, 512, 512, 512);
  wconv_k<<<cdiv(1024 * 512, B), B, 0, stream>>>(fc3w, wb3, 1024, 512, 512);
  wconv_k<<<cdiv(512 * 2048, B), B, 0, stream>>>(fc4w, wb4, 512, 2048, 2048);
  wconv_k<<<cdiv(128 * 512,  B), B, 0, stream>>>(fc5w, wb5, 128, 512, 512);
  pad_feats_k<<<cdiv(NCAM * NPTS * 8, B), B, 0, stream>>>(feats);

  // ---- 2) conv pyramid + sampling ----
  conv3x3_k<<<cdiv(NCAM * 8 * 512 * 512, B), B, 0, stream>>>(
      x, c1w, c1b, b1s, b1o, F0, 3, 8, 512, 512, 0);
  bilinear_sample_k<<<cdiv(NCAM * NPTS * 8, B), B, 0, stream>>>(
      F0, pts, feats, 8, 512, 512, 0, 1.0f);

  static const int CIN[6]  = {8, 16, 32, 64, 128, 256};
  static const int COUT[6] = {16, 32, 64, 128, 256, 512};
  float* cur = F0;
  float* nxt = F1;
  int chOff = 8;
  for (int s = 0; s < 6; ++s) {
    int Hs = 512 >> (s + 1);  // post-pool spatial
    int ci = CIN[s], cm = CIN[s], co = COUT[s];
    maxpool2_k<<<cdiv(NCAM * ci * Hs * Hs, B), B, 0, stream>>>(cur, PB, ci, Hs, Hs);
    conv3x3_k<<<cdiv(NCAM * cm * Hs * Hs, B), B, 0, stream>>>(
        PB, SC(s, 0), SC(s, 1), SC(s, 2), SC(s, 3), MB, ci, cm, Hs, Hs, 1);
    conv3x3_k<<<cdiv(NCAM * co * Hs * Hs, B), B, 0, stream>>>(
        MB, SC(s, 4), SC(s, 5), SC(s, 6), SC(s, 7), nxt, cm, co, Hs, Hs, 1);
    bilinear_sample_k<<<cdiv(NCAM * NPTS * co, B), B, 0, stream>>>(
        nxt, pts, feats, co, Hs, Hs, chOff, 1.0f / (float)(1 << (s + 1)));
    chOff += co;
    float* tmp = cur; cur = nxt; nxt = tmp;
  }

  // ---- 3) FC head via LDS-staged WMMA GEMMs (ping-pong R0 <-> R1) ----
  const int M = NCAM * NPTS;  // 65536
  __bf16* y1 = (__bf16*)R1;
  __bf16* y2 = feats;            // reuse R0 (feats dead after fc1)
  __bf16* y3 = (__bf16*)R1;      // reuse R1 (y1 dead after fc2)
  __bf16* pooled = (__bf16*)base;
  __bf16* y4 = (__bf16*)R1;
  __bf16* y5 = (__bf16*)base;

  gemm_bf16_wmma_lds<<<dim3(512 / GBN, M / GBM), B, 0, stream>>>(
      feats, wb1, fc1b, y1, M, 512, 1024, 1);
  gemm_bf16_wmma_lds<<<dim3(512 / GBN, M / GBM), B, 0, stream>>>(
      y1, wb2, fc2b, y2, M, 512, 512, 1);
  gemm_bf16_wmma_lds<<<dim3(1024 / GBN, M / GBM), B, 0, stream>>>(
      y2, wb3, fc3b, y3, M, 1024, 512, 1);
  pool_cams_k<<<cdiv(NPTS * 1024, B), B, 0, stream>>>(y3, pooled);
  gemm_bf16_wmma_lds<<<dim3(512 / GBN, NPTS / GBM), B, 0, stream>>>(
      pooled, wb4, fc4b, y4, NPTS, 512, 2048, 1);
  gemm_bf16_wmma_lds<<<dim3(128 / GBN, NPTS / GBM), B, 0, stream>>>(
      y4, wb5, fc5b, y5, NPTS, 128, 512, 1);
  fc6_k<<<cdiv(NPTS, B), B, 0, stream>>>(y5, fc6w, fc6b, (float*)d_out);
  #undef SC
}
